// FaceRecon_feat_24163486007682
// MI455X (gfx1250) — matrix operand order repaired and checked by reading, not compiled, not run on hardware
//
#include <hip/hip_runtime.h>
#include <math.h>
#include <float.h>
#include <stdint.h>

#define BSZ 2
#define NV 4096
#define KNBR 10
#define SUP 7

typedef __attribute__((ext_vector_type(16))) __bf16 v16bf;
typedef __attribute__((ext_vector_type(8)))  float  v8f;

// ---- A-fragment K map for V_WMMA_F32_16X16X32_BF16 (ISA 7.12.2, 16-bit A 16x32) ----
// VGPR v<4 : K = 2v + p + 8*half ; VGPR v>=4 : K = 16 + 2(v-4) + p + 8*half
__device__ __forceinline__ int amap_k(int e, int half) {
  int v = e >> 1, p = e & 1;
  return (v < 4) ? (v * 2 + p + half * 8) : (16 + (v - 4) * 2 + p + half * 8);
}

// ---- CDNA5 async global->LDS copy (ASYNCcnt-tracked), 16 bytes per lane ----
__device__ __forceinline__ void async_ld16(unsigned lds, const void* gaddr) {
  asm volatile("global_load_async_to_lds_b128 %0, %1, off"
               :: "v"(lds), "v"(gaddr) : "memory");
}
#define WAIT_ASYNCCNT(imm) asm volatile("s_wait_asynccnt " imm ::: "memory")

// ---------- normalize direction columns: d is (3, cols) row-major ----------
__global__ void norm_cols_k(const float* __restrict__ d, float* __restrict__ o, int cols) {
  int j = blockIdx.x * blockDim.x + threadIdx.x;
  if (j >= cols) return;
  float x = d[j], y = d[cols + j], z = d[2 * cols + j];
  float n = fmaxf(sqrtf(x * x + y * y + z * z), 1e-12f);
  o[j] = x / n; o[cols + j] = y / n; o[2 * cols + j] = z / n;
}

// ---------- kNN (top-kk excluding self), one thread per query point ----------
__global__ void knn_k(const float* __restrict__ verts, int* __restrict__ idx, int Vl, int kk) {
  int t = blockIdx.x * blockDim.x + threadIdx.x;
  if (t >= BSZ * Vl) return;
  int b = t / Vl, i = t - b * Vl;
  const float* vb = verts + (size_t)b * Vl * 3;
  float xi = vb[i * 3], yi = vb[i * 3 + 1], zi = vb[i * 3 + 2];
  const int kk1 = kk + 1;
  float bd[11]; int bi[11];
  for (int n = 0; n < 11; ++n) { bd[n] = FLT_MAX; bi[n] = 0; }
  for (int j = 0; j < Vl; ++j) {
    float xj = vb[j * 3], yj = vb[j * 3 + 1], zj = vb[j * 3 + 2];
    float d = xj * xj + yj * yj + zj * zj - 2.0f * (xi * xj + yi * yj + zi * zj);
    if (d < bd[kk1 - 1]) {
      int p = kk1 - 1;
      while (p > 0 && bd[p - 1] > d) { bd[p] = bd[p - 1]; bi[p] = bi[p - 1]; --p; }
      bd[p] = d; bi[p] = j;
    }
  }
  for (int n = 0; n < kk; ++n) idx[(size_t)t * kk + n] = bi[n + 1];
}

// ---------- conv_surface: one block per (b,v), 128 threads (one per kernel) ----------
__global__ void conv_surface_k(const int* __restrict__ idx, const float* __restrict__ verts,
                               const float* __restrict__ dn, float* __restrict__ fm0) {
  __shared__ float snd[KNBR][3];
  int bv = blockIdx.x;
  int b = bv / NV, v = bv - b * NV;
  int t = threadIdx.x;
  if (t < KNBR) {
    int j = idx[(size_t)bv * KNBR + t];
    const float* vb = verts + (size_t)b * NV * 3;
    float dx = vb[j * 3] - vb[v * 3];
    float dy = vb[j * 3 + 1] - vb[v * 3 + 1];
    float dz = vb[j * 3 + 2] - vb[v * 3 + 2];
    float n = fmaxf(sqrtf(dx * dx + dy * dy + dz * dz), 1e-12f);
    snd[t][0] = dx / n; snd[t][1] = dy / n; snd[t][2] = dz / n;
  }
  __syncthreads();
  int c = t;
  const int cols = SUP * 128;
  float acc = 0.f;
  #pragma unroll
  for (int s = 0; s < SUP; ++s) {
    int m = s * 128 + c;
    float dx = dn[m], dy = dn[cols + m], dz = dn[2 * cols + m];
    float mx = -FLT_MAX;
    #pragma unroll
    for (int n = 0; n < KNBR; ++n) {
      float th = fmaxf(dx * snd[n][0] + dy * snd[n][1] + dz * snd[n][2], 0.f);
      mx = fmaxf(mx, th);
    }
    acc += mx;
  }
  fm0[(size_t)bv * 128 + c] = fmaxf(acc, 0.f);
}

// ---------- fp32 -> bf16 conversions ----------
__global__ void cvt_bf16_k(const float* __restrict__ in, __bf16* __restrict__ out, long n) {
  long t = (long)blockIdx.x * blockDim.x + threadIdx.x;
  if (t < n) out[t] = (__bf16)in[t];
}
__global__ void transpose_cvt_k(const float* __restrict__ w, __bf16* __restrict__ wt, int K, int N) {
  long t = (long)blockIdx.x * blockDim.x + threadIdx.x;
  if (t >= (long)K * N) return;
  int k = (int)(t / N), n = (int)(t - (long)k * N);
  wt[(size_t)n * K + k] = (__bf16)w[t];
}

// ---------- WMMA GEMM: C(MxN) = A(MxK) * Bt(NxK)^T + bias ----------
// 128 threads = 4 waves; block tile 64x64; each wave: 16 rows x 64 cols (4 WMMA tiles).
// A/B tiles double-buffered in LDS via GLOBAL_LOAD_ASYNC_TO_LDS_B128 (ASYNCcnt).
__global__ __launch_bounds__(128) void gemm_bias_wmma_k(
    const __bf16* __restrict__ A, const __bf16* __restrict__ Bt,
    const float* __restrict__ bias, float* __restrict__ C,
    int M, int N, int K) {
  __shared__ __bf16 As[2][64][32];
  __shared__ __bf16 Bs[2][64][32];
  int tile_m = blockIdx.y * 64;
  int tile_n = blockIdx.x * 64;
  int tid = threadIdx.x, wave = tid >> 5, lane = tid & 31;
  int half = lane >> 4, lr = lane & 15;

  // Each thread async-copies two 16B chunks (8 bf16) of each 64x32 tile.
  int e0 = tid * 8;          // chunk 0 elements [e0, e0+8)
  int r0 = e0 >> 5, c0 = e0 & 31;
  int e1 = (tid + 128) * 8;  // chunk 1
  int r1 = e1 >> 5, c1 = e1 & 31;

  unsigned ldsA = (unsigned)(uintptr_t)&As[0][0][0];
  unsigned ldsB = (unsigned)(uintptr_t)&Bs[0][0][0];
  const unsigned bufBytes = 64u * 32u * 2u;  // 4KB per buffer

  auto stage = [&](int buf, int k0) {
    unsigned la = ldsA + (unsigned)buf * bufBytes;
    unsigned lb = ldsB + (unsigned)buf * bufBytes;
    async_ld16(la + (unsigned)e0 * 2u, &A [(size_t)(tile_m + r0) * K + k0 + c0]);
    async_ld16(la + (unsigned)e1 * 2u, &A [(size_t)(tile_m + r1) * K + k0 + c1]);
    async_ld16(lb + (unsigned)e0 * 2u, &Bt[(size_t)(tile_n + r0) * K + k0 + c0]);
    async_ld16(lb + (unsigned)e1 * 2u, &Bt[(size_t)(tile_n + r1) * K + k0 + c1]);
  };

  v8f acc[4];
  #pragma unroll
  for (int t4 = 0; t4 < 4; ++t4) acc[t4] = {};

  const int nk = K >> 5;
  stage(0, 0);
  for (int kt = 0; kt < nk; ++kt) {
    int buf = kt & 1;
    if (kt + 1 < nk) {
      stage(buf ^ 1, (kt + 1) << 5);   // prefetch next tile while computing
      WAIT_ASYNCCNT("0x4");            // in-order retire: current tile has landed
    } else {
      WAIT_ASYNCCNT("0x0");
    }
    __syncthreads();
    v16bf af;
    int am = wave * 16 + lr;
    #pragma unroll
    for (int e = 0; e < 16; ++e) af[e] = As[buf][am][amap_k(e, half)];
    #pragma unroll
    for (int t4 = 0; t4 < 4; ++t4) {
      v16bf bf;
      int bn = t4 * 16 + lr;
      #pragma unroll
      for (int e = 0; e < 16; ++e) bf[e] = Bs[buf][bn][amap_k(e, half)];
      acc[t4] = __builtin_amdgcn_wmma_f32_16x16x32_bf16(
          false, af, false, bf, (short)0, acc[t4], false, false);
    }
    __syncthreads();   // all waves done reading buf before it is restaged
  }
  #pragma unroll
  for (int t4 = 0; t4 < 4; ++t4) {
    int gn = tile_n + t4 * 16 + lr;
    float bvs = bias[gn];
    #pragma unroll
    for (int vv = 0; vv < 8; ++vv) {
      int gm = tile_m + wave * 16 + vv + 8 * half;
      C[(size_t)gm * N + gn] = acc[t4][vv] + bvs;
    }
  }
}

// ---------- conv_layer tail: f_center + sum_s max_n relu(theta) * f_support ----------
__global__ void conv_combine_k(const int* __restrict__ idx, const float* __restrict__ verts,
                               const float* __restrict__ fout, const float* __restrict__ dn,
                               float* __restrict__ fm, int Vl, int outc) {
  __shared__ float snd[KNBR][3];
  __shared__ int srow[KNBR];
  int bv = blockIdx.x;
  int b = bv / Vl, v = bv - b * Vl;
  int t = threadIdx.x;
  if (t < KNBR) {
    int j = idx[(size_t)bv * KNBR + t];
    srow[t] = b * Vl + j;
    const float* vb = verts + (size_t)b * Vl * 3;
    float dx = vb[j * 3] - vb[v * 3];
    float dy = vb[j * 3 + 1] - vb[v * 3 + 1];
    float dz = vb[j * 3 + 2] - vb[v * 3 + 2];
    float n = fmaxf(sqrtf(dx * dx + dy * dy + dz * dz), 1e-12f);
    snd[t][0] = dx / n; snd[t][1] = dy / n; snd[t][2] = dz / n;
  }
  __syncthreads();
  int c = t;
  int cols = SUP * outc;
  int N = (SUP + 1) * outc;
  float acc = fout[(size_t)bv * N + c];
  for (int s = 0; s < SUP; ++s) {
    int m = s * outc + c;
    float dx = dn[m], dy = dn[cols + m], dz = dn[2 * cols + m];
    int col = outc + m;
    float mx = -FLT_MAX;
    #pragma unroll
    for (int n = 0; n < KNBR; ++n) {
      float th = fmaxf(dx * snd[n][0] + dy * snd[n][1] + dz * snd[n][2], 0.f);
      float fs = fout[(size_t)srow[n] * N + col];
      mx = fmaxf(mx, th * fs);
    }
    acc += mx;
  }
  fm[(size_t)bv * outc + c] = acc;
}

// ---------- batchnorm: per-channel stats then normalize + relu ----------
__global__ void bn_stats_k(const float* __restrict__ x, float* __restrict__ mv, int Mrows, int C) {
  __shared__ float ss[256], sq[256];
  int ch = blockIdx.x, t = threadIdx.x;
  float s = 0.f, q = 0.f;
  for (int r = t; r < Mrows; r += 256) {
    float v = x[(size_t)r * C + ch];
    s += v; q += v * v;
  }
  ss[t] = s; sq[t] = q;
  __syncthreads();
  for (int o = 128; o > 0; o >>= 1) {
    if (t < o) { ss[t] += ss[t + o]; sq[t] += sq[t + o]; }
    __syncthreads();
  }
  if (t == 0) {
    float mean = ss[0] / (float)Mrows;
    mv[ch] = mean;
    mv[C + ch] = sq[0] / (float)Mrows - mean * mean;
  }
}
__global__ void bn_relu_k(float* __restrict__ x, const float* __restrict__ mv,
                          const float* __restrict__ g, const float* __restrict__ be,
                          int C, long total) {
  long t = (long)blockIdx.x * blockDim.x + threadIdx.x;
  if (t >= total) return;
  int ch = (int)(t % C);
  float y = (x[t] - mv[ch]) * rsqrtf(mv[C + ch] + 1e-5f) * g[ch] + be[ch];
  x[t] = fmaxf(y, 0.f);
}

// ---------- pool: deterministic 1/4 subsample + max over 4-NN features ----------
__device__ __forceinline__ int perm_map(int i, int Vl, unsigned seed) {
  unsigned a = (2654435761u ^ (seed * 0x9E3779B9u)) | 1u;   // odd => bijection mod 2^k
  unsigned c = seed * 0x85EBCA6Bu;
  return (int)((a * (unsigned)i + c) & (unsigned)(Vl - 1));
}
__global__ void pool_k(const float* __restrict__ verts, const float* __restrict__ fm,
                       float* __restrict__ vout, float* __restrict__ fpo,
                       int Vl, int C, unsigned seed) {
  __shared__ int srow[4];
  int Vq = Vl >> 2;
  int bo = blockIdx.x;
  int b = bo / Vq, oi = bo - b * Vq;
  int t = threadIdx.x;
  if (t == 0) {
    int src = perm_map(oi, Vl, seed);
    const float* vb = verts + (size_t)b * Vl * 3;
    float xi = vb[src * 3], yi = vb[src * 3 + 1], zi = vb[src * 3 + 2];
    float bd[5]; int bi[5];
    for (int n = 0; n < 5; ++n) { bd[n] = FLT_MAX; bi[n] = 0; }
    for (int j = 0; j < Vl; ++j) {
      float xj = vb[j * 3], yj = vb[j * 3 + 1], zj = vb[j * 3 + 2];
      float d = xj * xj + yj * yj + zj * zj - 2.f * (xi * xj + yi * yj + zi * zj);
      if (d < bd[4]) {
        int p = 4;
        while (p > 0 && bd[p - 1] > d) { bd[p] = bd[p - 1]; bi[p] = bi[p - 1]; --p; }
        bd[p] = d; bi[p] = j;
      }
    }
    for (int n = 0; n < 4; ++n) srow[n] = b * Vl + bi[n + 1];
    vout[(size_t)bo * 3]     = xi;
    vout[(size_t)bo * 3 + 1] = yi;
    vout[(size_t)bo * 3 + 2] = zi;
  }
  __syncthreads();
  float mx = -FLT_MAX;
  for (int n = 0; n < 4; ++n) mx = fmaxf(mx, fm[(size_t)srow[n] * C + t]);
  fpo[(size_t)bo * C + t] = mx;
}

// ---------- nearest source index for upsampling ----------
__global__ void nearest_k(const float* __restrict__ tgt, const float* __restrict__ src,
                          int* __restrict__ out, int Vt, int Vs) {
  int t = blockIdx.x * blockDim.x + threadIdx.x;
  if (t >= BSZ * Vt) return;
  int b = t / Vt, i = t - b * Vt;
  const float* tb = tgt + (size_t)b * Vt * 3;
  const float* sb = src + (size_t)b * Vs * 3;
  float xi = tb[i * 3], yi = tb[i * 3 + 1], zi = tb[i * 3 + 2];
  float best = FLT_MAX; int bj = 0;
  for (int j = 0; j < Vs; ++j) {
    float xj = sb[j * 3], yj = sb[j * 3 + 1], zj = sb[j * 3 + 2];
    float d = xj * xj + yj * yj + zj * zj - 2.f * (xi * xj + yi * yj + zi * zj);
    if (d < best) { best = d; bj = j; }
  }
  out[t] = bj;
}

// ---------- final concat [fm0 | fm1 | fm2@n1 | fm3@n1 | fm4@n2] ----------
__global__ void concat_k(const float* __restrict__ fm0, const float* __restrict__ fm1,
                         const float* __restrict__ fm2, const float* __restrict__ fm3,
                         const float* __restrict__ fm4,
                         const int* __restrict__ n1, const int* __restrict__ n2,
                         float* __restrict__ out) {
  long t = (long)blockIdx.x * blockDim.x + threadIdx.x;
  const long total = (long)BSZ * NV * 1280;
  if (t >= total) return;
  int c = (int)(t % 1280);
  long bv = t / 1280;
  int b = (int)(bv / NV);
  float r;
  if (c < 128)      r = fm0[bv * 128 + c];
  else if (c < 256) r = fm1[bv * 128 + (c - 128)];
  else if (c < 512) { long rr = (long)b * 1024 + n1[bv]; r = fm2[rr * 256 + (c - 256)]; }
  else if (c < 768) { long rr = (long)b * 1024 + n1[bv]; r = fm3[rr * 256 + (c - 512)]; }
  else              { long rr = (long)b * 256  + n2[bv]; r = fm4[rr * 512 + (c - 768)]; }
  out[t] = r;
}

extern "C" void kernel_launch(void* const* d_in, const int* in_sizes, int n_in,
                              void* d_out, int out_size, void* d_ws, size_t ws_size,
                              hipStream_t stream) {
  (void)in_sizes; (void)n_in; (void)out_size; (void)ws_size;
  const float* verts = (const float*)d_in[0];
  const float* dir0  = (const float*)d_in[1];
  const float* w1  = (const float*)d_in[2];
  const float* b1  = (const float*)d_in[3];
  const float* dir1 = (const float*)d_in[4];
  const float* g1  = (const float*)d_in[5];
  const float* be1 = (const float*)d_in[6];
  const float* w2  = (const float*)d_in[7];
  const float* b2  = (const float*)d_in[8];
  const float* dir2 = (const float*)d_in[9];
  const float* g2  = (const float*)d_in[10];
  const float* be2 = (const float*)d_in[11];
  const float* w3  = (const float*)d_in[12];
  const float* b3  = (const float*)d_in[13];
  const float* dir3 = (const float*)d_in[14];
  const float* g3  = (const float*)d_in[15];
  const float* be3 = (const float*)d_in[16];
  const float* w4  = (const float*)d_in[17];
  const float* b4  = (const float*)d_in[18];
  const float* dir4 = (const float*)d_in[19];

  char* p = (char*)d_ws;
  auto carve = [&](size_t bytes) -> void* {
    void* r = (void*)p;
    p += (bytes + 255) & ~(size_t)255;
    return r;
  };
  float* dn0 = (float*)carve(sizeof(float) * 3 * 896);
  float* dn1 = (float*)carve(sizeof(float) * 3 * 896);
  float* dn2 = (float*)carve(sizeof(float) * 3 * 1792);
  float* dn3 = (float*)carve(sizeof(float) * 3 * 1792);
  float* dn4 = (float*)carve(sizeof(float) * 3 * 3584);
  int* idx1 = (int*)carve(sizeof(int) * BSZ * NV * KNBR);
  int* idx2 = (int*)carve(sizeof(int) * BSZ * 1024 * KNBR);
  int* idx3 = (int*)carve(sizeof(int) * BSZ * 256 * KNBR);
  float* fm0 = (float*)carve(sizeof(float) * (size_t)BSZ * NV * 128);
  float* fm1 = (float*)carve(sizeof(float) * (size_t)BSZ * NV * 128);
  float* vp1 = (float*)carve(sizeof(float) * BSZ * 1024 * 3);
  float* fp1 = (float*)carve(sizeof(float) * (size_t)BSZ * 1024 * 128);
  float* fm2 = (float*)carve(sizeof(float) * (size_t)BSZ * 1024 * 256);
  float* fm3 = (float*)carve(sizeof(float) * (size_t)BSZ * 1024 * 256);
  float* vp2 = (float*)carve(sizeof(float) * BSZ * 256 * 3);
  float* fp2 = (float*)carve(sizeof(float) * (size_t)BSZ * 256 * 256);
  float* fm4 = (float*)carve(sizeof(float) * (size_t)BSZ * 256 * 512);
  int* n1 = (int*)carve(sizeof(int) * BSZ * NV);
  int* n2 = (int*)carve(sizeof(int) * BSZ * NV);
  float* mv = (float*)carve(sizeof(float) * 2 * 512);
  float* fout = (float*)carve(sizeof(float) * (size_t)8192 * 1024);
  __bf16* Abf = (__bf16*)carve(sizeof(__bf16) * (size_t)8192 * 128);
  __bf16* Wbf = (__bf16*)carve(sizeof(__bf16) * (size_t)4096 * 256);

  // normalized direction kernels
  norm_cols_k<<<(896 + 127) / 128, 128, 0, stream>>>(dir0, dn0, 896);
  norm_cols_k<<<(896 + 127) / 128, 128, 0, stream>>>(dir1, dn1, 896);
  norm_cols_k<<<(1792 + 127) / 128, 128, 0, stream>>>(dir2, dn2, 1792);
  norm_cols_k<<<(1792 + 127) / 128, 128, 0, stream>>>(dir3, dn3, 1792);
  norm_cols_k<<<(3584 + 127) / 128, 128, 0, stream>>>(dir4, dn4, 3584);

  // level 0
  knn_k<<<(BSZ * NV + 255) / 256, 256, 0, stream>>>(verts, idx1, NV, KNBR);
  conv_surface_k<<<BSZ * NV, 128, 0, stream>>>(idx1, verts, dn0, fm0);

  auto gemm = [&](const float* Af32, int M, int K, const float* W, const float* bias, int N) {
    long mk = (long)M * K, kn = (long)K * N;
    cvt_bf16_k<<<(int)((mk + 255) / 256), 256, 0, stream>>>(Af32, Abf, mk);
    transpose_cvt_k<<<(int)((kn + 255) / 256), 256, 0, stream>>>(W, Wbf, K, N);
    dim3 g(N / 64, M / 64);
    gemm_bias_wmma_k<<<g, 128, 0, stream>>>(Abf, Wbf, bias, fout, M, N, K);
  };

  // layer 1: (8192x128)x(128x1024)
  gemm(fm0, BSZ * NV, 128, w1, b1, 1024);
  conv_combine_k<<<BSZ * NV, 128, 0, stream>>>(idx1, verts, fout, dn1, fm1, NV, 128);
  bn_stats_k<<<128, 256, 0, stream>>>(fm1, mv, BSZ * NV, 128);
  bn_relu_k<<<(int)(((long)BSZ * NV * 128 + 255) / 256), 256, 0, stream>>>(
      fm1, mv, g1, be1, 128, (long)BSZ * NV * 128);

  // pool 1 (4096 -> 1024) + level-1 kNN
  pool_k<<<BSZ * 1024, 128, 0, stream>>>(verts, fm1, vp1, fp1, NV, 128, 1u);
  knn_k<<<(BSZ * 1024 + 255) / 256, 256, 0, stream>>>(vp1, idx2, 1024, KNBR);

  // layer 2: (2048x128)x(128x2048)
  gemm(fp1, BSZ * 1024, 128, w2, b2, 2048);
  conv_combine_k<<<BSZ * 1024, 256, 0, stream>>>(idx2, vp1, fout, dn2, fm2, 1024, 256);
  bn_stats_k<<<256, 256, 0, stream>>>(fm2, mv, BSZ * 1024, 256);
  bn_relu_k<<<(int)(((long)BSZ * 1024 * 256 + 255) / 256), 256, 0, stream>>>(
      fm2, mv, g2, be2, 256, (long)BSZ * 1024 * 256);

  // layer 3: (2048x256)x(256x2048)
  gemm(fm2, BSZ * 1024, 256, w3, b3, 2048);
  conv_combine_k<<<BSZ * 1024, 256, 0, stream>>>(idx2, vp1, fout, dn3, fm3, 1024, 256);
  bn_stats_k<<<256, 256, 0, stream>>>(fm3, mv, BSZ * 1024, 256);
  bn_relu_k<<<(int)(((long)BSZ * 1024 * 256 + 255) / 256), 256, 0, stream>>>(
      fm3, mv, g3, be3, 256, (long)BSZ * 1024 * 256);

  // pool 2 (1024 -> 256) + level-2 kNN
  pool_k<<<BSZ * 256, 256, 0, stream>>>(vp1, fm3, vp2, fp2, 1024, 256, 2u);
  knn_k<<<(BSZ * 256 + 255) / 256, 256, 0, stream>>>(vp2, idx3, 256, KNBR);

  // layer 4: (512x256)x(256x4096), no BN/relu
  gemm(fp2, BSZ * 256, 256, w4, b4, 4096);
  conv_combine_k<<<BSZ * 256, 512, 0, stream>>>(idx3, vp2, fout, dn4, fm4, 256, 512);

  // upsample indices + concat
  nearest_k<<<(BSZ * NV + 255) / 256, 256, 0, stream>>>(verts, vp1, n1, NV, 1024);
  nearest_k<<<(BSZ * NV + 255) / 256, 256, 0, stream>>>(verts, vp2, n2, NV, 256);
  long total = (long)BSZ * NV * 1280;
  concat_k<<<(int)((total + 255) / 256), 256, 0, stream>>>(
      fm0, fm1, fm2, fm3, fm4, n1, n2, (float*)d_out);
}
